// Gtransformerblock_44375602102406
// MI455X (gfx1250) — matrix operand order repaired and checked by reading, not compiled, not run on hardware
//
#include <hip/hip_runtime.h>
#include <hip/hip_bf16.h>

// MI455X / gfx1250 graph-transformer attention.
// Workspace layout (needs 12 MB): Qf16[8192*256] | Kf16[8192*256] | Vt_f16[256*8192]

typedef _Float16 v16h __attribute__((ext_vector_type(16)));
typedef _Float16 v8h  __attribute__((ext_vector_type(8)));
typedef float    v8f  __attribute__((ext_vector_type(8)));
typedef float    f4   __attribute__((ext_vector_type(4)));

#define N_NODES 8192
#define DIM     256

union Frag { v16h v; v8h h[2]; _Float16 e[16]; };

static __device__ __forceinline__ v8f wmma_f16(v16h a, v16h b, v8f c) {
  return __builtin_amdgcn_wmma_f32_16x16x32_f16(false, a, false, b, (short)0, c, false, false);
}

// ---------------------------------------------------------------------------
// QKV projection: out = h @ W^T, converted to f16.
// grid (512, 3): y=0 -> Q (pre-scaled by 1/sqrt(256)), y=1 -> K, y=2 -> V^T.
// Block = 128 threads = 4 waves; wave w owns output cols [w*64, w*64+64).
// ---------------------------------------------------------------------------
__global__ __launch_bounds__(128) void qkv_kernel(
    const float* __restrict__ h,  const float* __restrict__ Wq,
    const float* __restrict__ Wk, const float* __restrict__ Wv,
    _Float16* __restrict__ Qf, _Float16* __restrict__ Kf,
    _Float16* __restrict__ Vt)
{
  const int wave  = threadIdx.x >> 5;
  const int lane  = threadIdx.x & 31;
  const int half  = lane >> 4;
  const int mrow  = lane & 15;
  const int rbase = blockIdx.x * 16;
  const int which = blockIdx.y;

  const float* W  = (which == 0) ? Wq : ((which == 1) ? Wk : Wv);
  const float scl = (which == 0) ? 0.0625f : 1.0f;   // fold 1/sqrt(IN_DIM) into Q

  v8f acc[4] = {};
  const float* hrow = h + (size_t)(rbase + mrow) * DIM;

  for (int c = 0; c < 8; ++c) {
    const int k0 = c * 32;
    // A fragment (16-bit 16x32): halves[0..7]=K k0+half*8.., halves[8..15]=K k0+16+half*8..
    Frag af;
    {
      f4 a0 = *(const f4*)(hrow + k0 + half * 8);
      f4 a1 = *(const f4*)(hrow + k0 + half * 8 + 4);
      f4 a2 = *(const f4*)(hrow + k0 + 16 + half * 8);
      f4 a3 = *(const f4*)(hrow + k0 + 16 + half * 8 + 4);
      #pragma unroll
      for (int i = 0; i < 4; ++i) {
        af.e[i]      = (_Float16)(a0[i] * scl);
        af.e[4 + i]  = (_Float16)(a1[i] * scl);
        af.e[8 + i]  = (_Float16)(a2[i] * scl);
        af.e[12 + i] = (_Float16)(a3[i] * scl);
      }
    }
    #pragma unroll
    for (int nt = 0; nt < 4; ++nt) {
      // B fragment (16-bit 32x16): lane n = col n, halves[0..7]=K k0+half*8.., halves[8..15]=K k0+16+half*8..
      const float* wrow = W + (size_t)(wave * 64 + nt * 16 + mrow) * DIM + k0 + half * 8;
      Frag bf;
      f4 b0 = *(const f4*)(wrow);
      f4 b1 = *(const f4*)(wrow + 4);
      f4 b2 = *(const f4*)(wrow + 16);
      f4 b3 = *(const f4*)(wrow + 20);
      #pragma unroll
      for (int i = 0; i < 4; ++i) {
        bf.e[i]      = (_Float16)b0[i];
        bf.e[4 + i]  = (_Float16)b1[i];
        bf.e[8 + i]  = (_Float16)b2[i];
        bf.e[12 + i] = (_Float16)b3[i];
      }
      acc[nt] = wmma_f16(af.v, bf.v, acc[nt]);
    }
  }

  #pragma unroll
  for (int nt = 0; nt < 4; ++nt) {
    const int col = wave * 64 + nt * 16 + mrow;
    #pragma unroll
    for (int v = 0; v < 8; ++v) {
      const int row = rbase + v + half * 8;
      _Float16 x = (_Float16)acc[nt][v];
      if (which == 0)      Qf[(size_t)row * DIM + col] = x;
      else if (which == 1) Kf[(size_t)row * DIM + col] = x;
      else                 Vt[(size_t)col * N_NODES + row] = x;   // transposed V
    }
  }
}

// ---------------------------------------------------------------------------
// Fused masked attention: one workgroup (4 waves) per 16 query rows.
// Waves are INDEPENDENT over keys (wave w handles key tiles (t*4+w)*32):
// no barriers in the main loop; the C->A layout transpose of the score tile
// uses a private per-wave LDS tile (same-wave DS ordering only).
// Single-pass softmax (no max subtraction; masked scores are O(1)-bounded).
// ---------------------------------------------------------------------------
__global__ __launch_bounds__(128) void attn_kernel(
    const float* __restrict__ adj,
    const _Float16* __restrict__ Qf, const _Float16* __restrict__ Kf,
    const _Float16* __restrict__ Vt, float* __restrict__ out)
{
  __shared__ float stile[4 * 16 * 32];    // per-wave score transpose tiles (8 KB)
  __shared__ float redbuf[2 * 16 * DIM];  // numerator pairwise reduction (32 KB)
  __shared__ float dred[4 * 16];          // per-wave denominator partials

  const int wave  = threadIdx.x >> 5;
  const int lane  = threadIdx.x & 31;
  const int half  = lane >> 4;
  const int mrow  = lane & 15;
  const int qbase = blockIdx.x * 16;

  // Full 256-dim Q A-fragments for these 16 rows (scale pre-folded).
  Frag qf[8];
  #pragma unroll
  for (int c = 0; c < 8; ++c) {
    const _Float16* p = Qf + (size_t)(qbase + mrow) * DIM + c * 32 + half * 16;
    qf[c].h[0] = *(const v8h*)(p);
    qf[c].h[1] = *(const v8h*)(p + 8);
  }

  v8f   oacc[16] = {};                    // 16 rows x 256 cols f32 numerator
  float dacc     = 0.f;
  float* mybuf   = stile + wave * 512;
  const float* arow = adj + (size_t)(qbase + mrow) * N_NODES;

  for (int t = 0; t < N_NODES / 128; ++t) {
    const int j0 = (t * 4 + wave) * 32;   // this wave's 32-key tile

    // ---- S = Q @ K^T over full 256 dims: 16 WMMAs ----
    v8f s0 = {}, s1 = {};
    #pragma unroll
    for (int c = 0; c < 8; ++c) {
      v16h kf0 = *(const v16h*)(Kf + (size_t)(j0 + mrow) * DIM + c * 32 + half * 16);
      v16h kf1 = *(const v16h*)(Kf + (size_t)(j0 + 16 + mrow) * DIM + c * 32 + half * 16);
      s0 = wmma_f16(qf[c].v, kf0, s0);
      s1 = wmma_f16(qf[c].v, kf1, s1);
    }

    // ---- C-layout -> A-layout transpose through private LDS tile ----
    #pragma unroll
    for (int v = 0; v < 8; ++v) {
      const int m = v + half * 8;
      mybuf[m * 32 + mrow]      = s0[v];
      mybuf[m * 32 + 16 + mrow] = s1[v];
    }
    const float* b = mybuf + mrow * 32 + half * 16;  // same-wave RAW: dscnt wait only
    f4 lo_a = *(const f4*)(b);
    f4 lo_b = *(const f4*)(b + 4);
    f4 hi_a = *(const f4*)(b + 8);
    f4 hi_b = *(const f4*)(b + 12);

    // ---- adjacency mask (stream-once -> non-temporal) ----
    const float* ap = arow + j0 + half * 16;
    f4 ad_lo_a = __builtin_nontemporal_load((const f4*)(ap));
    f4 ad_lo_b = __builtin_nontemporal_load((const f4*)(ap + 4));
    f4 ad_hi_a = __builtin_nontemporal_load((const f4*)(ap + 8));
    f4 ad_hi_b = __builtin_nontemporal_load((const f4*)(ap + 12));
    if (j0 + 160 < N_NODES) __builtin_prefetch(ap + 128, 0, 0);

    // ---- exp(masked score); build P A-fragment; accumulate denominator ----
    Frag pf;
    float rs = 0.f;
    #pragma unroll
    for (int i = 0; i < 4; ++i) {
      float p0 = __expf(lo_a[i] * ad_lo_a[i]);
      float p1 = __expf(lo_b[i] * ad_lo_b[i]);
      float p2 = __expf(hi_a[i] * ad_hi_a[i]);
      float p3 = __expf(hi_b[i] * ad_hi_b[i]);
      rs += p0 + p1 + p2 + p3;
      pf.e[i]      = (_Float16)p0;
      pf.e[4 + i]  = (_Float16)p1;
      pf.e[8 + i]  = (_Float16)p2;
      pf.e[12 + i] = (_Float16)p3;
    }
    dacc += rs;

    // ---- P @ V across all 256 output cols: 16 WMMAs ----
    #pragma unroll
    for (int nt = 0; nt < 16; ++nt) {
      v16h vf = *(const v16h*)(Vt + (size_t)(nt * 16 + mrow) * N_NODES + j0 + half * 16);
      oacc[nt] = wmma_f16(pf.v, vf, oacc[nt]);
    }
  }

  // ---- deterministic cross-wave reduction (no atomics) ----
  float dall = dacc + __shfl_xor(dacc, 16, 32);   // lane L holds this wave's denom[L%16]
  if (lane < 16) dred[wave * 16 + mrow] = dall;

  float* nb = redbuf + (size_t)(wave & 1) * 16 * DIM;
  if (wave < 2) {                                  // waves 0,1 initialize two buffers
    #pragma unroll
    for (int nt = 0; nt < 16; ++nt)
      #pragma unroll
      for (int v = 0; v < 8; ++v)
        nb[(v + half * 8) * DIM + nt * 16 + mrow] = oacc[nt][v];
  }
  __syncthreads();
  if (wave >= 2) {                                 // waves 2,3 add into disjoint buffers
    #pragma unroll
    for (int nt = 0; nt < 16; ++nt)
      #pragma unroll
      for (int v = 0; v < 8; ++v)
        nb[(v + half * 8) * DIM + nt * 16 + mrow] += oacc[nt][v];
  }
  __syncthreads();

  // ---- normalize + store: wave w handles rows [w*4, w*4+4) ----
  for (int i = 0; i < 32; ++i) {
    const int idx = wave * (4 * DIM) + i * 32 + lane;
    const int r   = idx >> 8;
    const int cl  = idx & (DIM - 1);
    const float den = (dred[r] + dred[32 + r]) + (dred[16 + r] + dred[48 + r]);
    const float num = redbuf[r * DIM + cl] + redbuf[16 * DIM + r * DIM + cl];
    out[(size_t)(qbase + r) * DIM + cl] = num / den;
  }
}

// ---------------------------------------------------------------------------
extern "C" void kernel_launch(void* const* d_in, const int* in_sizes, int n_in,
                              void* d_out, int out_size, void* d_ws, size_t ws_size,
                              hipStream_t stream) {
  const float* adj = (const float*)d_in[0];
  const float* h   = (const float*)d_in[1];
  const float* Wq  = (const float*)d_in[2];
  const float* Wk  = (const float*)d_in[3];
  const float* Wv  = (const float*)d_in[4];
  float* out = (float*)d_out;

  _Float16* Qf = (_Float16*)d_ws;                    //  4 MB
  _Float16* Kf = Qf + (size_t)N_NODES * DIM;         //  4 MB
  _Float16* Vt = Kf + (size_t)N_NODES * DIM;         //  4 MB (transposed V)

  dim3 gq(N_NODES / 16, 3, 1);
  qkv_kernel<<<gq, 128, 0, stream>>>(h, Wq, Wk, Wv, Qf, Kf, Vt);
  attn_kernel<<<N_NODES / 16, 128, 0, stream>>>(adj, Qf, Kf, Vt, out);
}
